// AdaptiveContextRouter_32865089749378
// MI455X (gfx1250) — matrix operand order repaired and checked by reading, not compiled, not run on hardware
//
#include <hip/hip_runtime.h>
#include <hip/hip_bf16.h>
#include <stdint.h>

// ---------------------------------------------------------------------------
// Types for CDNA5 WMMA (wave32)
// ---------------------------------------------------------------------------
typedef __attribute__((ext_vector_type(16))) __bf16 v16bf;
typedef __attribute__((ext_vector_type(8)))  float  v8f;

union FragAB {
    v16bf v;
    uint4 q[2];
};

#define T_TOK   (4 * 4096)   // B*S = 16384 tokens
#define D_MODEL 1024
#define POOL_N  4096
#define HID_N   256
#define K_SEL   256

// ---------------------------------------------------------------------------
// fp32 -> bf16 (round to nearest even) conversion
// ---------------------------------------------------------------------------
__global__ void acr_f32_to_bf16(const float* __restrict__ in,
                                unsigned short* __restrict__ out, int n) {
    int i = blockIdx.x * blockDim.x + threadIdx.x;
    if (i < n) {
        uint32_t u = __float_as_uint(in[i]);
        uint32_t r = u + 0x7FFFu + ((u >> 16) & 1u);
        out[i] = (unsigned short)(r >> 16);
    }
}

// ---------------------------------------------------------------------------
// bf16 WMMA GEMM:  out[M x N] = A[M x K] * W[N x K]^T + bias (opt. ReLU)
// Block: 256 threads = 8 waves (wave32). Wave tile = 32x64 (2x4 WMMA frags).
// Block tile = 64 (M) x 256 (N). M%64==0, N%256==0, K%32==0 assumed.
// Per K-step: 4 A-loads + 8 B-loads (b128) feed 8 WMMAs (1.5 loads/WMMA).
// ---------------------------------------------------------------------------
__global__ __launch_bounds__(256) void acr_gemm_bf16_wmma(
    const unsigned short* __restrict__ A,   // bf16 bits, [M x K]
    const unsigned short* __restrict__ W,   // bf16 bits, [N x K]
    const float* __restrict__ bias,         // [N]
    float* __restrict__ out,                // [M x N]
    int M, int N, int K, int do_relu) {

    const int lane     = threadIdx.x & 31;
    const int wave     = threadIdx.x >> 5;
    const int wm       = wave >> 2;          // 0..1
    const int wn       = wave & 3;           // 0..3
    const int m0       = blockIdx.x * 64 + wm * 32;
    const int n0       = blockIdx.y * 256 + wn * 64;
    const int laneHi   = (lane & 16) ? 1 : 0;
    const int l15      = lane & 15;

    v8f acc[2][4];
    #pragma unroll
    for (int mi = 0; mi < 2; ++mi)
        #pragma unroll
        for (int ni = 0; ni < 4; ++ni)
            #pragma unroll
            for (int r = 0; r < 8; ++r) acc[mi][ni][r] = 0.0f;

    for (int k0 = 0; k0 < K; k0 += 32) {
        FragAB a[2], b[4];
        // A fragment: 16x32, lane holds row (l15), K-chunks:
        //   VGPR0-3: K = k0 + (laneHi?8:0) .. +7 ; VGPR4-7: K = +16 more
        const int ka = k0 + (laneHi ? 8 : 0);
        #pragma unroll
        for (int mi = 0; mi < 2; ++mi) {
            const unsigned short* pa = A + (size_t)(m0 + 16 * mi + l15) * K;
            a[mi].q[0] = *(const uint4*)(pa + ka);
            a[mi].q[1] = *(const uint4*)(pa + ka + 16);
        }
        // B fragment: 32x16, lane holds column N=l15,
        //   VGPR0-3: K = k0 + (laneHi?8:0) .. +7 ; VGPR4-7: K = +16 more (W is [N x K])
        const int kb = k0 + (laneHi ? 8 : 0);
        #pragma unroll
        for (int ni = 0; ni < 4; ++ni) {
            const unsigned short* pb = W + (size_t)(n0 + 16 * ni + l15) * K + kb;
            b[ni].q[0] = *(const uint4*)(pb);
            b[ni].q[1] = *(const uint4*)(pb + 16);
        }
        #pragma unroll
        for (int mi = 0; mi < 2; ++mi)
            #pragma unroll
            for (int ni = 0; ni < 4; ++ni)
                acc[mi][ni] = __builtin_amdgcn_wmma_f32_16x16x32_bf16(
                    false, a[mi].v, false, b[ni].v,
                    (short)0, acc[mi][ni], false, false);
    }

    // Epilogue: C/D layout -> VGPR r holds M = r + (laneHi?8:0), N = l15
    #pragma unroll
    for (int ni = 0; ni < 4; ++ni) {
        const int col = n0 + 16 * ni + l15;
        const float bv = bias[col];
        #pragma unroll
        for (int mi = 0; mi < 2; ++mi) {
            #pragma unroll
            for (int r = 0; r < 8; ++r) {
                const int row = m0 + 16 * mi + r + (laneHi ? 8 : 0);
                float v = acc[mi][ni][r] + bv;
                if (do_relu) v = fmaxf(v, 0.0f);
                out[(size_t)row * N + col] = v;
            }
        }
    }
}

// ---------------------------------------------------------------------------
// complexity head: sigmoid(h . c2_w + c2_b) -> k_values (trunc to int32)
// ---------------------------------------------------------------------------
__global__ void acr_complexity(const float* __restrict__ h,
                               const float* __restrict__ c2w,
                               const float* __restrict__ c2b,
                               int* __restrict__ k_ws,
                               int* __restrict__ k_out, int n) {
    int t = blockIdx.x * blockDim.x + threadIdx.x;
    if (t >= n) return;
    const float* hr = h + (size_t)t * HID_N;
    float z = c2b[0];
    for (int i = 0; i < HID_N; ++i) z += hr[i] * c2w[i];
    float s = 1.0f / (1.0f + __expf(-z));
    int k = (int)(32.0f + s * 224.0f);   // K_MIN + c*(K_MAX-K_MIN), trunc
    k_ws[t]  = k;
    k_out[t] = k;
}

// ---------------------------------------------------------------------------
// Per-token top-256 of 4096 (bitonic sort in LDS, descending, tie -> low idx)
// then recompute only the 256 selected wq dot-products and apply the k-mask.
// One workgroup (256 threads) per token.
// ---------------------------------------------------------------------------
__global__ __launch_bounds__(256) void acr_topk_gather(
    const float* __restrict__ sel_scores,  // [T x POOL] (already in d_out)
    const float* __restrict__ x,           // [T x D] fp32 (original input)
    const float* __restrict__ wq_w,        // [POOL x D]
    const float* __restrict__ wq_b,        // [POOL]
    const int*  __restrict__ kvals,        // [T]
    int*   __restrict__ idx_out,           // [T x 256]
    float* __restrict__ pw_out) {          // [T x 256]

    __shared__ unsigned long long keys[POOL_N];
    __shared__ float xrow[D_MODEL];

    const int t   = blockIdx.x;
    const int tid = threadIdx.x;

    // Build sortable keys: (monotone-float << 32) | (POOL-1-idx) so that
    // descending u64 order == descending score, low index first on ties.
    const float* row = sel_scores + (size_t)t * POOL_N;
    for (int i = tid; i < POOL_N; i += 256) {
        uint32_t u = __float_as_uint(row[i]);
        u = (u & 0x80000000u) ? ~u : (u | 0x80000000u);
        keys[i] = ((unsigned long long)u << 32) | (unsigned)(POOL_N - 1 - i);
    }
    __syncthreads();

    // Bitonic sort, descending
    for (unsigned k = 2; k <= (unsigned)POOL_N; k <<= 1) {
        for (unsigned j = k >> 1; j > 0; j >>= 1) {
            for (int i = tid; i < POOL_N; i += 256) {
                unsigned p = (unsigned)i ^ j;
                if (p > (unsigned)i) {
                    bool up = ((i & k) == 0);
                    unsigned long long ai = keys[i], ap = keys[p];
                    if (up ? (ai < ap) : (ai > ap)) {
                        keys[i] = ap;
                        keys[p] = ai;
                    }
                }
            }
            __syncthreads();
        }
    }

    // Stage the x row for the gathered dot products
    for (int i = tid; i < D_MODEL; i += 256) xrow[i] = x[(size_t)t * D_MODEL + i];
    __syncthreads();

    // Thread tid owns rank `tid`
    unsigned long long key = keys[tid];
    int idx = POOL_N - 1 - (int)(unsigned)(key & 0xFFFFFFFFu);

    const float4* wr = (const float4*)(wq_w + (size_t)idx * D_MODEL);
    const float4* xr = (const float4*)xrow;
    float s = wq_b[idx];
    #pragma unroll 4
    for (int d = 0; d < D_MODEL / 4; ++d) {
        float4 wv = wr[d];
        float4 xv = xr[d];
        s += wv.x * xv.x + wv.y * xv.y + wv.z * xv.z + wv.w * xv.w;
    }
    if (tid >= kvals[t]) s = 0.0f;

    idx_out[(size_t)t * K_SEL + tid] = idx;
    pw_out [(size_t)t * K_SEL + tid] = s;
}

// ---------------------------------------------------------------------------
// Launcher
// ---------------------------------------------------------------------------
extern "C" void kernel_launch(void* const* d_in, const int* in_sizes, int n_in,
                              void* d_out, int out_size, void* d_ws, size_t ws_size,
                              hipStream_t stream) {
    const float* x     = (const float*)d_in[0];
    const float* sel_w = (const float*)d_in[1];
    const float* sel_b = (const float*)d_in[2];
    const float* wq_w  = (const float*)d_in[3];
    const float* wq_b  = (const float*)d_in[4];
    const float* c1_w  = (const float*)d_in[5];
    const float* c1_b  = (const float*)d_in[6];
    const float* c2_w  = (const float*)d_in[7];
    const float* c2_b  = (const float*)d_in[8];

    // Workspace layout
    char* ws = (char*)d_ws;
    size_t off = 0;
    unsigned short* x_bf    = (unsigned short*)(ws + off); off += (size_t)T_TOK * D_MODEL * 2;
    unsigned short* selw_bf = (unsigned short*)(ws + off); off += (size_t)POOL_N * D_MODEL * 2;
    unsigned short* c1_bf   = (unsigned short*)(ws + off); off += (size_t)HID_N * D_MODEL * 2;
    float*          h_buf   = (float*)(ws + off);          off += (size_t)T_TOK * HID_N * 4;
    int*            k_ws    = (int*)(ws + off);            off += (size_t)T_TOK * 4;
    (void)ws_size; (void)in_sizes; (void)n_in; (void)out_size;

    // Output layout (tuple flattened: indices, pattern_weights, sel_scores, k)
    char* ob = (char*)d_out;
    int*   idx_out = (int*)ob;
    float* pw_out  = (float*)(ob + (size_t)T_TOK * K_SEL * 4);
    float* ss_out  = (float*)(ob + (size_t)T_TOK * K_SEL * 4 * 2);
    int*   k_out   = (int*)(ob + (size_t)T_TOK * K_SEL * 4 * 2 + (size_t)T_TOK * POOL_N * 4);

    // 1. Conversions to bf16
    {
        int n = T_TOK * D_MODEL;
        acr_f32_to_bf16<<<(n + 255) / 256, 256, 0, stream>>>(x, x_bf, n);
        n = POOL_N * D_MODEL;
        acr_f32_to_bf16<<<(n + 255) / 256, 256, 0, stream>>>(sel_w, selw_bf, n);
        n = HID_N * D_MODEL;
        acr_f32_to_bf16<<<(n + 255) / 256, 256, 0, stream>>>(c1_w, c1_bf, n);
    }

    // 2. Selection-score GEMM (WMMA bf16) -> directly into d_out sel_scores
    acr_gemm_bf16_wmma<<<dim3(T_TOK / 64, POOL_N / 256), 256, 0, stream>>>(
        x_bf, selw_bf, sel_b, ss_out, T_TOK, POOL_N, D_MODEL, 0);

    // 3. Complexity hidden layer GEMM (WMMA bf16, fused ReLU)
    acr_gemm_bf16_wmma<<<dim3(T_TOK / 64, HID_N / 256), 256, 0, stream>>>(
        x_bf, c1_bf, c1_b, h_buf, T_TOK, HID_N, D_MODEL, 1);

    // 4. Sigmoid head -> k_values
    acr_complexity<<<(T_TOK + 255) / 256, 256, 0, stream>>>(
        h_buf, c2_w, c2_b, k_ws, k_out, T_TOK);

    // 5. Top-k + gathered wq dot products + mask
    acr_topk_gather<<<T_TOK, 256, 0, stream>>>(
        ss_out, x, wq_w, wq_b, k_ws, idx_out, pw_out);
}
